// AttentionFuse_46840913330535
// MI455X (gfx1250) — compile-verified, matrix operand order checked by reading, not run on hardware
//
#include <hip/hip_runtime.h>

#define D_MODEL 128
#define NTOK    4096      // T*H*W
#define BATCH   2
#define HW      1024
#define KT      32        // keys per flash-attention tile
#define SCALE   0.08838834764831845f   // 1/sqrt(128)

typedef __attribute__((ext_vector_type(16))) _Float16 v16h;
typedef __attribute__((ext_vector_type(8)))  float    v8f;
typedef __attribute__((ext_vector_type(4))) unsigned int u32x4;
typedef __attribute__((ext_vector_type(8)))  int      i32x8;
typedef __attribute__((ext_vector_type(4)))  int      i32x4;

#if defined(__has_builtin)
#  if __has_builtin(__builtin_amdgcn_tensor_load_to_lds) && \
      __has_builtin(__builtin_amdgcn_s_wait_tensorcnt)
#    define HAVE_TDM 1
#  endif
#endif
#ifndef HAVE_TDM
#  define HAVE_TDM 0
#endif

union V16 { v16h h; u32x4 q[2]; };

// Load a 16-half A/B fragment as two 16-byte chunks (ISA layout: per-lane,
// halves [0..7] then [16..23] relative to the lane's K sub-block base).
__device__ __forceinline__ v16h load_frag_h16(const _Float16* p0, const _Float16* p1) {
  V16 x;
  x.q[0] = *(const u32x4*)p0;
  x.q[1] = *(const u32x4*)p1;
  return x.h;
}

#if HAVE_TDM
// Issue one TDM 2-D tile load: dim0 contiguous elements (2B each), dim1 rows
// with stride0 elements between rows. Tile == described tensor (no clipping).
// D# packing per CDNA5 ISA ch.8 (group0 128b, group1 256b, groups 2/3 zero).
// clang-23 / therock-10.0 builtin: 6 args (g0, g1, g2, g3, extra, cpol).
__device__ __forceinline__ void tdm_load_2d(unsigned lds_addr, const void* gptr,
                                            unsigned dim0, unsigned dim1,
                                            unsigned long long stride0) {
  const unsigned long long ga = (unsigned long long)(size_t)gptr;
  u32x4 g0 = { 1u,                                   // count=1, user mode
               lds_addr,                             // LDS byte address
               (unsigned)(ga & 0xFFFFFFFFu),         // global_addr[31:0]
               ((unsigned)((ga >> 32) & 0x01FFFFFFu))
                 | 0x80000000u };                    // global_addr[56:32] | type=2
  i32x8 g1 = {
    (int)0x00010000u,                                          // data_size=2B
    (int)((dim0 & 0xFFFFu) << 16),                             // tensor_dim0 lo16
    (int)(((dim0 >> 16) & 0xFFFFu) | ((dim1 & 0xFFFFu) << 16)),// td0 hi / td1 lo
    (int)(((dim1 >> 16) & 0xFFFFu) | ((dim0 & 0xFFFFu) << 16)),// td1 hi / tile_dim0
    (int)(dim1 & 0xFFFFu),                                     // tile_dim1, tile_dim2=0
    (int)(unsigned)(stride0 & 0xFFFFFFFFull),                  // tensor_dim0_stride lo
    (int)(unsigned)((stride0 >> 32) & 0xFFFFull),              // stride hi, td1_stride=0
    0
  };
  i32x4 gz4 = { 0, 0, 0, 0 };
  i32x8 gz8 = { 0, 0, 0, 0, 0, 0, 0, 0 };
  __builtin_amdgcn_tensor_load_to_lds(g0, g1, gz4, gz4, gz8, 0);
}
#endif

// ---------------------------------------------------------------------------
// Kernel 1: fused permute + QKV projection.
//   q = (x Wq^T + bq) * scale   -> Qws[b, n, d]  (f16, scale folded in)
//   k =  x Wk^T + bk            -> Kws[b, n, d]  (f16)
//   v =  x Wv^T + bv            -> Vt [b, d, n]  (f16, transposed for PV WMMA)
// ---------------------------------------------------------------------------
__global__ __launch_bounds__(256)
void qkv_project_kernel(const float* __restrict__ query,
                        const float* __restrict__ key_value,
                        const float* __restrict__ Wq, const float* __restrict__ bq,
                        const float* __restrict__ Wk, const float* __restrict__ bk,
                        const float* __restrict__ Wv, const float* __restrict__ bv,
                        _Float16* __restrict__ qout,
                        _Float16* __restrict__ kout,
                        _Float16* __restrict__ vtout)
{
  const int id  = blockIdx.x;
  const int mat = id / (BATCH * (NTOK / 16));
  const int rem = id % (BATCH * (NTOK / 16));
  const int b   = rem / (NTOK / 16);
  const int nt  = rem % (NTOK / 16);
  const int n0  = nt * 16;
  const int t   = n0 >> 10;          // 16-token tile never crosses a T slice
  const int hw0 = n0 & (HW - 1);

  const float* src  = (mat == 0) ? query : key_value;
  const float* W    = (mat == 0) ? Wq : (mat == 1) ? Wk : Wv;
  const float* bias = (mat == 0) ? bq : (mat == 1) ? bk : bv;

  __shared__ __attribute__((aligned(16))) _Float16 sin[16 * D_MODEL];

  const int tid = threadIdx.x;
  {
    const int m  = tid & 15;      // token within tile (consecutive hw)
    const int cb = tid >> 4;      // channel group
    const size_t base = ((size_t)(t * BATCH + b) * D_MODEL) * HW + hw0 + m;
#pragma unroll
    for (int i = 0; i < 8; ++i) {
      const int c = cb + 16 * i;
      sin[m * D_MODEL + c] = (_Float16)src[base + (size_t)c * HW];
    }
  }
  __syncthreads();

  const int wave = tid >> 5;
  const int lane = tid & 31;
  const int ll   = lane & 15;
  const int hi   = lane >> 4;
  const int koff = hi * 8;

  const int d = wave * 16 + ll;              // output channel (B-fragment col)
  const float* wrow = W + (size_t)d * D_MODEL;

  v8f acc = {};
#pragma unroll
  for (int cc = 0; cc < 4; ++cc) {           // K = 128 in 4 x 32 steps
    const int c0 = cc * 32 + koff;
    v16h a = load_frag_h16(&sin[ll * D_MODEL + c0], &sin[ll * D_MODEL + c0 + 16]);
    V16 bf;
#pragma unroll
    for (int i = 0; i < 8; ++i) {
      bf.h[i]     = (_Float16)wrow[c0 + i];
      bf.h[i + 8] = (_Float16)wrow[c0 + 16 + i];
    }
    acc = __builtin_amdgcn_wmma_f32_16x16x32_f16(false, a, false, bf.h,
                                                 (short)0, acc, false, false);
  }

  const float bd = bias[d];
#pragma unroll
  for (int r = 0; r < 8; ++r) {
    float v = acc[r] + bd;
    if (mat == 0) v *= SCALE;                // fold softmax scale into Q
    const int row = n0 + r + 8 * hi;         // C/D layout: M = r + 8*laneHi
    if (mat == 0)      qout [((size_t)b * NTOK + row) * D_MODEL + d] = (_Float16)v;
    else if (mat == 1) kout [((size_t)b * NTOK + row) * D_MODEL + d] = (_Float16)v;
    else               vtout[((size_t)b * D_MODEL + d) * NTOK + row] = (_Float16)v;
  }
}

// ---------------------------------------------------------------------------
// Kernel 2: flash attention. 8 waves/block, each wave owns 16 query rows.
// K/V tiles are staged into LDS once per block by the Tensor Data Mover,
// double-buffered so the DMA of tile t+1 overlaps compute of tile t.
// ---------------------------------------------------------------------------
__global__ __launch_bounds__(256)
void flash_attn_kernel(const _Float16* __restrict__ Q,    // [B,N,C] pre-scaled
                       const _Float16* __restrict__ K,    // [B,N,C]
                       const _Float16* __restrict__ Vt,   // [B,C,N]
                       float* __restrict__ out)           // [B,N,C] f32
{
  const int b    = blockIdx.x / (NTOK / 128);
  const int mblk = blockIdx.x % (NTOK / 128);
  const int tid  = threadIdx.x;
  const int wave = tid >> 5;
  const int lane = tid & 31;
  const int ll   = lane & 15;
  const int hi   = lane >> 4;
  const int koff = hi * 8;

  const int m0 = mblk * 128 + wave * 16;

  __shared__ __attribute__((aligned(16))) _Float16 Kbuf[2][KT * D_MODEL]; // [tok][c]
  __shared__ __attribute__((aligned(16))) _Float16 Vbuf[2][D_MODEL * KT]; // [d][tok]
  __shared__ __attribute__((aligned(16))) _Float16 pbuf[8][16 * KT];

  const _Float16* Kb = K  + (size_t)b * NTOK * D_MODEL;
  const _Float16* Vb = Vt + (size_t)b * D_MODEL * NTOK;

  // Resident Q fragments: A-layout, row = ll, K sub-block per lane half.
  const _Float16* qrow = Q + ((size_t)b * NTOK + m0 + ll) * D_MODEL;
  v16h qf[4];
#pragma unroll
  for (int cc = 0; cc < 4; ++cc) {
    const int c0 = cc * 32 + koff;
    qf[cc] = load_frag_h16(qrow + c0, qrow + c0 + 16);
  }

  v8f acc[8];
  float mrow[8], lrow[8];
#pragma unroll
  for (int j = 0; j < 8; ++j) acc[j] = (v8f){};
#pragma unroll
  for (int r = 0; r < 8; ++r) { mrow[r] = -3.0e38f; lrow[r] = 0.0f; }

  const int ntiles = NTOK / KT;

#if HAVE_TDM
  // Generic shared pointers carry the LDS byte offset in their low 32 bits.
  const unsigned kLds[2] = { (unsigned)(size_t)(const void*)&Kbuf[0][0],
                             (unsigned)(size_t)(const void*)&Kbuf[1][0] };
  const unsigned vLds[2] = { (unsigned)(size_t)(const void*)&Vbuf[0][0],
                             (unsigned)(size_t)(const void*)&Vbuf[1][0] };
  if (wave == 0) {   // prologue: DMA tile 0 into buffer 0
    tdm_load_2d(kLds[0], Kb, D_MODEL, KT, D_MODEL);
    tdm_load_2d(vLds[0], Vb, KT, D_MODEL, NTOK);
  }
#endif

  for (int kt = 0; kt < ntiles; ++kt) {
    const int cur = kt & 1;

    __syncthreads();   // everyone done reading the buffer about to be refilled
#if HAVE_TDM
    if (wave == 0) {
      if (kt + 1 < ntiles) {
        const int nx = (kt + 1) & 1;
        tdm_load_2d(kLds[nx], Kb + (size_t)(kt + 1) * KT * D_MODEL,
                    D_MODEL, KT, D_MODEL);
        tdm_load_2d(vLds[nx], Vb + (size_t)(kt + 1) * KT,
                    KT, D_MODEL, NTOK);
        __builtin_amdgcn_s_wait_tensorcnt(2);   // tile kt complete (in-order)
      } else {
        __builtin_amdgcn_s_wait_tensorcnt(0);
      }
    }
#else
    // Fallback: cooperative synchronous staging of tile kt (whole block).
    {
      const u32x4* ksrc = (const u32x4*)(Kb + (size_t)kt * KT * D_MODEL);
      u32x4* kdst = (u32x4*)&Kbuf[cur][0];
      kdst[tid * 2]     = ksrc[tid * 2];
      kdst[tid * 2 + 1] = ksrc[tid * 2 + 1];
      const int vrow = tid >> 1;
      const int vh   = (tid & 1) * 16;
      const _Float16* vsrc = Vb + (size_t)vrow * NTOK + (size_t)kt * KT + vh;
      *(u32x4*)&Vbuf[cur][vrow * KT + vh]     = *(const u32x4*)(vsrc);
      *(u32x4*)&Vbuf[cur][vrow * KT + vh + 8] = *(const u32x4*)(vsrc + 8);
    }
#endif
    __syncthreads();   // tile kt visible in LDS to all waves

    // S = Q K^T : two 16x16 n-tiles, K=128 in 4 steps each (from LDS).
    v8f s[2] = {(v8f){}, (v8f){}};
#pragma unroll
    for (int ntile = 0; ntile < 2; ++ntile) {
      const _Float16* krow = &Kbuf[cur][(ntile * 16 + ll) * D_MODEL + koff];
#pragma unroll
      for (int cc = 0; cc < 4; ++cc) {
        v16h kf = load_frag_h16(krow + cc * 32, krow + cc * 32 + 16);
        s[ntile] = __builtin_amdgcn_wmma_f32_16x16x32_f16(false, qf[cc], false, kf,
                                                          (short)0, s[ntile], false, false);
      }
    }

    // Online softmax. Row stats live per (VGPR index r, lane half); the xor
    // reductions with masks 1,2,4,8 stay inside each 16-lane half.
#pragma unroll
    for (int r = 0; r < 8; ++r) {
      float v = fmaxf(s[0][r], s[1][r]);
      v = fmaxf(v, __shfl_xor(v, 1, 32));
      v = fmaxf(v, __shfl_xor(v, 2, 32));
      v = fmaxf(v, __shfl_xor(v, 4, 32));
      v = fmaxf(v, __shfl_xor(v, 8, 32));
      const float mn = fmaxf(mrow[r], v);
      const float f  = __expf(mrow[r] - mn);
      mrow[r] = mn;
      lrow[r] *= f;
#pragma unroll
      for (int j = 0; j < 8; ++j) acc[j][r] *= f;
      const float p0 = __expf(s[0][r] - mn);
      const float p1 = __expf(s[1][r] - mn);
      float ps = p0 + p1;
      ps += __shfl_xor(ps, 1, 32);
      ps += __shfl_xor(ps, 2, 32);
      ps += __shfl_xor(ps, 4, 32);
      ps += __shfl_xor(ps, 8, 32);
      lrow[r] += ps;
      // Stash P (C/D layout) into per-wave LDS, row-major [16][32] f16.
      const int prow = (r + 8 * hi) * KT;
      pbuf[wave][prow + ll]      = (_Float16)p0;
      pbuf[wave][prow + 16 + ll] = (_Float16)p1;
    }
    __syncthreads();
    // Re-read P in A-fragment layout (16-byte ds loads).
    v16h pf = load_frag_h16(&pbuf[wave][ll * KT + koff],
                            &pbuf[wave][ll * KT + koff + 16]);
    __syncthreads();

    // O += P V : 8 d-tiles, K = 32 key tokens (V rows contiguous in LDS).
#pragma unroll
    for (int j = 0; j < 8; ++j) {
      const _Float16* vrow = &Vbuf[cur][(j * 16 + ll) * KT + koff];
      v16h vf = load_frag_h16(vrow, vrow + 16);
      acc[j] = __builtin_amdgcn_wmma_f32_16x16x32_f16(false, pf, false, vf,
                                                      (short)0, acc[j], false, false);
    }
  }

  // Epilogue: divide by row sums, write f32 output.
#pragma unroll
  for (int r = 0; r < 8; ++r) {
    const float inv = 1.0f / lrow[r];
    const int row = m0 + r + 8 * hi;
    float* orow = out + ((size_t)b * NTOK + row) * D_MODEL;
#pragma unroll
    for (int j = 0; j < 8; ++j) orow[j * 16 + ll] = acc[j][r] * inv;
  }
}

// ---------------------------------------------------------------------------
extern "C" void kernel_launch(void* const* d_in, const int* in_sizes, int n_in,
                              void* d_out, int out_size, void* d_ws, size_t ws_size,
                              hipStream_t stream) {
  const float* query = (const float*)d_in[0];
  const float* keyv  = (const float*)d_in[1];
  const float* Wq    = (const float*)d_in[2];
  const float* bq    = (const float*)d_in[3];
  const float* Wk    = (const float*)d_in[4];
  const float* bk    = (const float*)d_in[5];
  const float* Wv    = (const float*)d_in[6];
  const float* bv    = (const float*)d_in[7];
  float* out = (float*)d_out;

  _Float16* ws = (_Float16*)d_ws;
  const size_t mat_elems = (size_t)BATCH * NTOK * D_MODEL;   // 1M halves each
  _Float16* Qws  = ws;
  _Float16* Kws  = ws + mat_elems;
  _Float16* Vtws = ws + 2 * mat_elems;

  qkv_project_kernel<<<3 * BATCH * (NTOK / 16), 256, 0, stream>>>(
      query, keyv, Wq, bq, Wk, bk, Wv, bv, Qws, Kws, Vtws);

  flash_attn_kernel<<<BATCH * (NTOK / 128), 256, 0, stream>>>(
      Qws, Kws, Vtws, out);
}
